// EfficientSelfAttention_81793357185264
// MI455X (gfx1250) — matrix operand order repaired and checked by reading, not compiled, not run on hardware
//
#include <hip/hip_runtime.h>

// ---------------------------------------------------------------- types
typedef __attribute__((ext_vector_type(16))) __bf16 v16bf;
typedef __attribute__((ext_vector_type(8)))  float  v8f;

union FragU { v16bf v; uint4 q[2]; };
union Ld16U { uint4 q[2]; __bf16 e[16]; };

__device__ __forceinline__ v8f wmma_bf16(v16bf a, v16bf b, v8f c) {
    return __builtin_amdgcn_wmma_f32_16x16x32_bf16(false, a, false, b, (short)0, c, false, false);
}

// ---------------- CDNA5 async global->LDS DMA (ASYNCcnt-tracked) ----------------
// INST_OFFSET is added to BOTH the global and the LDS address (ISA 15.18.3),
// so one base pair covers multiple 16B chunks via offset:N.
template <int OFF>
__device__ __forceinline__ void async_ld_b128(unsigned lds_addr, unsigned long long gaddr) {
    asm volatile("global_load_async_to_lds_b128 %0, %1, off offset:%2"
                 :: "v"(lds_addr), "v"(gaddr), "n"(OFF)
                 : "memory");
}

__device__ __forceinline__ void wait_async0() {
#if __has_builtin(__builtin_amdgcn_s_wait_asynccnt)
    __builtin_amdgcn_s_wait_asynccnt(0);
#else
    asm volatile("s_wait_asynccnt 0" ::: "memory");
#endif
}

// Generic LDS pointer -> 32-bit LDS offset (ISA 10.2: LDS aperture truncates addr[31:0])
__device__ __forceinline__ unsigned lds_off(const void* p) {
    return (unsigned)(unsigned long long)(size_t)p;
}

// A fragment 16x32 bf16 (M rows, K cols). tile stored row-major [row][k], stride in elems.
// lane<16: K {0..7,16..23}; lane>=16: K {8..15,24..31}; row = rowBase + (lane&15)
__device__ __forceinline__ v16bf load_a_frag(const __bf16* tile, int stride, int rowBase,
                                             int kBase, int lane) {
    FragU f;
    const __bf16* p = tile + (rowBase + (lane & 15)) * stride + kBase + ((lane < 16) ? 0 : 8);
    f.q[0] = *(const uint4*)(p);
    f.q[1] = *(const uint4*)(p + 16);
    return f.v;
}

// B fragment 32x16 bf16 (K rows, N cols). tile stored [col][k] (k contiguous).
// lane<16: K {0..7,16..23}; lane>=16: K {8..15,24..31}; col = colBase + (lane&15)
__device__ __forceinline__ v16bf load_b_frag(const __bf16* tile, int stride, int colBase,
                                             int kBase, int lane) {
    FragU f;
    const __bf16* p = tile + (colBase + (lane & 15)) * stride + kBase + ((lane < 16) ? 0 : 8);
    f.q[0] = *(const uint4*)(p);
    f.q[1] = *(const uint4*)(p + 16);
    return f.v;
}

// ---------------------------------------------------------------- small utility kernels
__global__ __launch_bounds__(256) void conv_bf16_kernel(const float* __restrict__ src,
                                                        unsigned short* __restrict__ dst,
                                                        size_t n) {
    size_t i = (size_t)blockIdx.x * 256 + threadIdx.x;
    if (i < n) ((__bf16*)dst)[i] = (__bf16)src[i];
}

__global__ __launch_bounds__(256) void zero_kernel(float* __restrict__ p, size_t n) {
    size_t i = (size_t)blockIdx.x * 256 + threadIdx.x;
    if (i < n) p[i] = 0.0f;
}

// Fused weight: rows 0..1023 = w_f @ Wq (per head), 1024..2047 = w_f @ Wk, 2048..3071 = Wv
__global__ __launch_bounds__(256) void fuse_w_kernel(const float* __restrict__ w_qkv,
                                                     const float* __restrict__ w_f,
                                                     unsigned short* __restrict__ Wb) {
    size_t idx = (size_t)blockIdx.x * 256 + threadIdx.x;   // f*1024 + e
    if (idx >= (size_t)3072 * 1024) return;
    int f = (int)(idx >> 10), e = (int)(idx & 1023);
    float acc;
    if (f < 2048) {
        int seg = f >> 10;            // 0 = q, 1 = k
        int h = (f & 1023) >> 6, p = f & 63;
        acc = 0.0f;
        const float* wq = w_qkv + ((size_t)seg * 1024 + h * 64) * 1024 + e;
        const float* wf = w_f + p * 64;
        for (int d = 0; d < 64; ++d) acc += wf[d] * wq[(size_t)d * 1024];
    } else {
        acc = w_qkv[(size_t)f * 1024 + e];
    }
    ((__bf16*)Wb)[idx] = (__bf16)acc;
}

// ---------------------------------------------------------------- qkv GEMM
// M=32768 (m = s*8+b), N=3072, K=1024. Block tile 128x128, K-step 32, 8 waves.
// Staging via GLOBAL_LOAD_ASYNC_TO_LDS_B128 double-buffered.
__global__ __launch_bounds__(256) void qkv_gemm_kernel(const unsigned short* __restrict__ Xb_,
                                                       const unsigned short* __restrict__ Wb_,
                                                       const float* __restrict__ b_f,
                                                       unsigned short* __restrict__ qp_,
                                                       unsigned short* __restrict__ kp_,
                                                       unsigned short* __restrict__ vv_) {
    constexpr int K = 1024, AS = 40;
    const __bf16* Xb = (const __bf16*)Xb_;
    const __bf16* Wb = (const __bf16*)Wb_;
    __bf16* qp = (__bf16*)qp_; __bf16* kp = (__bf16*)kp_; __bf16* vv = (__bf16*)vv_;

    __shared__ __align__(16) __bf16 At[2][128 * AS];
    __shared__ __align__(16) __bf16 Bt[2][128 * AS];

    const int m0 = blockIdx.x * 128, f0 = blockIdx.y * 128;
    const int tid = threadIdx.x, lane = tid & 31, wave = tid >> 5;
    const int wm = (wave >> 1) * 32, wn = (wave & 1) * 64;

    v8f acc[2][4] = {};

    const int lr = tid >> 1, lh = tid & 1;     // 2 threads per row, 32B each

    auto stage_async = [&](int buf, int k0) {
        unsigned ldsA = lds_off(&At[buf][lr * AS + lh * 16]);
        unsigned long long gA =
            (unsigned long long)(size_t)(Xb + (size_t)(m0 + lr) * K + k0 + lh * 16);
        async_ld_b128<0>(ldsA, gA);
        async_ld_b128<16>(ldsA, gA);
        unsigned ldsB = lds_off(&Bt[buf][lr * AS + lh * 16]);
        unsigned long long gB =
            (unsigned long long)(size_t)(Wb + (size_t)(f0 + lr) * K + k0 + lh * 16);
        async_ld_b128<0>(ldsB, gB);
        async_ld_b128<16>(ldsB, gB);
    };

    stage_async(0, 0);
    const int nk = K / 32;
    for (int kt = 0; kt < nk; ++kt) {
        wait_async0();              // my async writes into buf(kt) are complete
        __syncthreads();            // everyone's are; everyone done reading buf^1
        const int buf = kt & 1;
        if (kt + 1 < nk) stage_async(buf ^ 1, (kt + 1) * 32);
        v16bf a[2], b[4];
        for (int mt = 0; mt < 2; ++mt) a[mt] = load_a_frag(At[buf], AS, wm + mt * 16, 0, lane);
        for (int nt = 0; nt < 4; ++nt) b[nt] = load_b_frag(Bt[buf], AS, wn + nt * 16, 0, lane);
        for (int mt = 0; mt < 2; ++mt)
            for (int nt = 0; nt < 4; ++nt)
                acc[mt][nt] = wmma_bf16(a[mt], b[nt], acc[mt][nt]);
    }

    // epilogue: decode f -> (q'|k'|v); block's column range never straddles segments.
    for (int mt = 0; mt < 2; ++mt) {
        const int rb = wm + mt * 16 + ((lane >= 16) ? 8 : 0);
        for (int nt = 0; nt < 4; ++nt) {
            const int f = f0 + wn + nt * 16 + (lane & 15);
            for (int j = 0; j < 8; ++j) {
                const int m = m0 + rb + j;
                const int s = m >> 3, bb = m & 7;
                float val = acc[mt][nt][j];
                if (f < 1024) {
                    int h = f >> 6, p = f & 63;
                    val += b_f[p]; val = val > 0.f ? val : 0.f;
                    qp[(((size_t)(h * 8 + bb)) * 4096 + s) * 64 + p] = (__bf16)val;
                } else if (f < 2048) {
                    int f2 = f - 1024, h = f2 >> 6, p = f2 & 63;
                    val += b_f[p]; val = val > 0.f ? val : 0.f;
                    kp[(((size_t)(h * 8 + bb)) * 4096 + s) * 64 + p] = (__bf16)val;
                } else {
                    int f2 = f - 2048, h = f2 >> 6, d = f2 & 63;
                    vv[(((size_t)(h * 8 + bb)) * 4096 + s) * 64 + d] = (__bf16)val;
                }
            }
        }
    }
}

// ---------------------------------------------------------------- kv = k'^T v  (split-K, atomics)
// grid (128 hb, 8 kchunk), 128 threads (4 waves, each 32x32 of the 64x64 output)
// Transposing stage must go through VGPRs (async DMA cannot transpose).
__global__ __launch_bounds__(128) void kv_kernel(const unsigned short* __restrict__ kp_,
                                                 const unsigned short* __restrict__ vv_,
                                                 float* __restrict__ kv,
                                                 float* __restrict__ ksum) {
    const __bf16* kp = (const __bf16*)kp_;
    const __bf16* vv = (const __bf16*)vv_;
    __shared__ __align__(16) __bf16 KtT[64 * 40];   // [p][s]
    __shared__ __align__(16) __bf16 VtT[64 * 40];   // [d][s]

    const int hb = blockIdx.x, kc = blockIdx.y;
    const int sbeg = kc * 512;
    const int tid = threadIdx.x, lane = tid & 31, wave = tid >> 5;
    const int wp = (wave >> 1) * 32, wd = (wave & 1) * 32;

    const __bf16* kbase = kp + (size_t)hb * 4096 * 64;
    const __bf16* vbase = vv + (size_t)hb * 4096 * 64;

    v8f acc[2][2] = {};
    float ks = 0.0f;

    const int sr = tid >> 2, c0 = (tid & 3) * 16;   // 32 s-rows, 4 threads/row
    for (int it = 0; it < 16; ++it) {
        const int s0 = sbeg + it * 32;
        Ld16U kl, vl;
        const __bf16* krow = kbase + (size_t)(s0 + sr) * 64 + c0;
        const __bf16* vrow = vbase + (size_t)(s0 + sr) * 64 + c0;
        kl.q[0] = *(const uint4*)(krow);  kl.q[1] = *(const uint4*)(krow + 8);
        vl.q[0] = *(const uint4*)(vrow);  vl.q[1] = *(const uint4*)(vrow + 8);
        __syncthreads();                   // previous iteration's readers done
        for (int e = 0; e < 16; ++e) {
            KtT[(c0 + e) * 40 + sr] = kl.e[e];
            VtT[(c0 + e) * 40 + sr] = vl.e[e];
        }
        __syncthreads();
        v16bf a[2], b[2];
        for (int mt = 0; mt < 2; ++mt) a[mt] = load_a_frag(KtT, 40, wp + mt * 16, 0, lane);
        for (int nt = 0; nt < 2; ++nt) b[nt] = load_b_frag(VtT, 40, wd + nt * 16, 0, lane);
        for (int mt = 0; mt < 2; ++mt)
            for (int nt = 0; nt < 2; ++nt)
                acc[mt][nt] = wmma_bf16(a[mt], b[nt], acc[mt][nt]);
        if (tid < 64) {
            float s = 0.f;
            for (int e = 0; e < 32; ++e) s += (float)KtT[tid * 40 + e];
            ks += s;
        }
    }
    for (int mt = 0; mt < 2; ++mt) {
        const int pb = wp + mt * 16 + ((lane >= 16) ? 8 : 0);
        for (int nt = 0; nt < 2; ++nt) {
            const int d = wd + nt * 16 + (lane & 15);
            for (int j = 0; j < 8; ++j)
                atomicAdd(&kv[((size_t)hb * 64 + pb + j) * 64 + d], acc[mt][nt][j]);
        }
    }
    if (tid < 64) atomicAdd(&ksum[hb * 64 + tid], ks);
}

// ---------------------------------------------------------------- attn = (q' kv) / (q'.k_sum)
// grid (32 s-tiles, 128 hb), 256 threads; wave w -> 16 rows, full 64 cols, K=64 (2 steps)
__global__ __launch_bounds__(256) void attn_kernel(const unsigned short* __restrict__ qp_,
                                                   const float* __restrict__ kv,
                                                   const float* __restrict__ ksum,
                                                   unsigned short* __restrict__ attn_) {
    const __bf16* qp = (const __bf16*)qp_;
    __bf16* attn = (__bf16*)attn_;
    __shared__ __align__(16) __bf16 kvT[64 * 72];   // [d][p]
    __shared__ __align__(16) __bf16 Qt[128 * 72];   // [r][p]
    __shared__ float ksumS[64];
    __shared__ float rnorm[128];

    const int hb = blockIdx.y, h = hb >> 3, bb = hb & 7;
    const int s0 = blockIdx.x * 128;
    const int tid = threadIdx.x, lane = tid & 31, wave = tid >> 5;

    {   // stage 128 q' rows (64 bf16 each) via async DMA: 64B per thread
        const int r = tid >> 1, c0 = (tid & 1) * 32;
        unsigned ldsQ = lds_off(&Qt[r * 72 + c0]);
        unsigned long long gQ =
            (unsigned long long)(size_t)(qp + ((size_t)hb * 4096 + s0 + r) * 64 + c0);
        async_ld_b128<0>(ldsQ, gQ);
        async_ld_b128<16>(ldsQ, gQ);
        async_ld_b128<32>(ldsQ, gQ);
        async_ld_b128<48>(ldsQ, gQ);
    }
    {   // stage kv (f32) transposed into bf16 [d][p]
        const int p = tid >> 2, d0 = (tid & 3) * 16;
        const float* kr = kv + ((size_t)hb * 64 + p) * 64 + d0;
        for (int e = 0; e < 16; ++e) kvT[(d0 + e) * 72 + p] = (__bf16)kr[e];
    }
    if (tid < 64) ksumS[tid] = ksum[hb * 64 + tid];
    wait_async0();
    __syncthreads();

    if (tid < 128) {
        float sum = 0.f;
        for (int p = 0; p < 64; ++p) sum += (float)Qt[tid * 72 + p] * ksumS[p];
        rnorm[tid] = 1.0f / (sum + 1e-8f);
    }

    v8f acc[4] = {};
    for (int kstep = 0; kstep < 2; ++kstep) {
        v16bf a = load_a_frag(Qt, 72, wave * 16, kstep * 32, lane);
        for (int nt = 0; nt < 4; ++nt) {
            v16bf b = load_b_frag(kvT, 72, nt * 16, kstep * 32, lane);
            acc[nt] = wmma_bf16(a, b, acc[nt]);
        }
    }
    __syncthreads();   // rnorm visible

    for (int nt = 0; nt < 4; ++nt) {
        const int d = nt * 16 + (lane & 15);
        const int rb = wave * 16 + ((lane >= 16) ? 8 : 0);
        for (int j = 0; j < 8; ++j) {
            const int r = rb + j;
            const float val = acc[nt][j] * rnorm[r];
            attn[((size_t)(s0 + r) * 8 + bb) * 1024 + h * 64 + d] = (__bf16)val;
        }
    }
}

// ---------------------------------------------------------------- output projection
// M=32768, N=1024, K=1024, f32 out + bias; async-staged double buffer
__global__ __launch_bounds__(256) void out_gemm_kernel(const unsigned short* __restrict__ Ab_,
                                                       const unsigned short* __restrict__ Wob_,
                                                       const float* __restrict__ b_out,
                                                       float* __restrict__ out) {
    constexpr int K = 1024, AS = 40;
    const __bf16* Ab = (const __bf16*)Ab_;
    const __bf16* Wob = (const __bf16*)Wob_;

    __shared__ __align__(16) __bf16 At[2][128 * AS];
    __shared__ __align__(16) __bf16 Bt[2][128 * AS];

    const int m0 = blockIdx.x * 128, f0 = blockIdx.y * 128;
    const int tid = threadIdx.x, lane = tid & 31, wave = tid >> 5;
    const int wm = (wave >> 1) * 32, wn = (wave & 1) * 64;

    v8f acc[2][4] = {};
    const int lr = tid >> 1, lh = tid & 1;

    auto stage_async = [&](int buf, int k0) {
        unsigned ldsA = lds_off(&At[buf][lr * AS + lh * 16]);
        unsigned long long gA =
            (unsigned long long)(size_t)(Ab + (size_t)(m0 + lr) * K + k0 + lh * 16);
        async_ld_b128<0>(ldsA, gA);
        async_ld_b128<16>(ldsA, gA);
        unsigned ldsB = lds_off(&Bt[buf][lr * AS + lh * 16]);
        unsigned long long gB =
            (unsigned long long)(size_t)(Wob + (size_t)(f0 + lr) * K + k0 + lh * 16);
        async_ld_b128<0>(ldsB, gB);
        async_ld_b128<16>(ldsB, gB);
    };

    stage_async(0, 0);
    const int nk = K / 32;
    for (int kt = 0; kt < nk; ++kt) {
        wait_async0();
        __syncthreads();
        const int buf = kt & 1;
        if (kt + 1 < nk) stage_async(buf ^ 1, (kt + 1) * 32);
        v16bf a[2], b[4];
        for (int mt = 0; mt < 2; ++mt) a[mt] = load_a_frag(At[buf], AS, wm + mt * 16, 0, lane);
        for (int nt = 0; nt < 4; ++nt) b[nt] = load_b_frag(Bt[buf], AS, wn + nt * 16, 0, lane);
        for (int mt = 0; mt < 2; ++mt)
            for (int nt = 0; nt < 4; ++nt)
                acc[mt][nt] = wmma_bf16(a[mt], b[nt], acc[mt][nt]);
    }

    for (int mt = 0; mt < 2; ++mt) {
        const int rb = wm + mt * 16 + ((lane >= 16) ? 8 : 0);
        for (int nt = 0; nt < 4; ++nt) {
            const int f = f0 + wn + nt * 16 + (lane & 15);
            const float bias = b_out[f];
            for (int j = 0; j < 8; ++j) {
                const int m = m0 + rb + j;
                out[(size_t)m * 1024 + f] = acc[mt][nt][j] + bias;
            }
        }
    }
}

// ---------------------------------------------------------------- launch
extern "C" void kernel_launch(void* const* d_in, const int* in_sizes, int n_in,
                              void* d_out, int out_size, void* d_ws, size_t ws_size,
                              hipStream_t stream) {
    (void)in_sizes; (void)n_in; (void)out_size; (void)ws_size;
    const float* x     = (const float*)d_in[0];
    const float* w_qkv = (const float*)d_in[1];
    const float* w_out = (const float*)d_in[2];
    const float* b_out = (const float*)d_in[3];
    const float* w_f   = (const float*)d_in[4];
    const float* b_f   = (const float*)d_in[5];

    size_t off = 0;
    char* base = (char*)d_ws;
    auto take = [&](size_t bytes) -> void* {
        void* p = base + off;
        off += (bytes + 255) & ~(size_t)255;
        return p;
    };
    unsigned short* Xb   = (unsigned short*)take((size_t)32768 * 1024 * 2);
    unsigned short* Wb   = (unsigned short*)take((size_t)3072 * 1024 * 2);
    unsigned short* Wob  = (unsigned short*)take((size_t)1024 * 1024 * 2);
    unsigned short* qp   = (unsigned short*)take((size_t)128 * 4096 * 64 * 2);
    unsigned short* kp   = (unsigned short*)take((size_t)128 * 4096 * 64 * 2);
    unsigned short* vv   = (unsigned short*)take((size_t)128 * 4096 * 64 * 2);
    float*          kv   = (float*)take((size_t)128 * 64 * 64 * 4);
    float*          ksum = (float*)take((size_t)128 * 64 * 4);
    unsigned short* attn = (unsigned short*)take((size_t)32768 * 1024 * 2);

    conv_bf16_kernel<<<131072, 256, 0, stream>>>(x, Xb, (size_t)32768 * 1024);
    conv_bf16_kernel<<<4096, 256, 0, stream>>>(w_out, Wob, (size_t)1024 * 1024);
    fuse_w_kernel<<<12288, 256, 0, stream>>>(w_qkv, w_f, Wb);
    zero_kernel<<<2048, 256, 0, stream>>>(kv, (size_t)524288);
    zero_kernel<<<32, 256, 0, stream>>>(ksum, (size_t)8192);

    qkv_gemm_kernel<<<dim3(256, 24), 256, 0, stream>>>(Xb, Wb, b_f, qp, kp, vv);
    kv_kernel<<<dim3(128, 8), 128, 0, stream>>>(kp, vv, kv, ksum);
    attn_kernel<<<dim3(32, 128), 256, 0, stream>>>(qp, kv, ksum, attn);
    out_gemm_kernel<<<dim3(256, 8), 256, 0, stream>>>(attn, Wob, b_out, (float*)d_out);
}
